// MCSVD_58686433133018
// MI455X (gfx1250) — matrix operand order repaired and checked by reading, not compiled, not run on hardware
//
#include <hip/hip_runtime.h>
#include <hip/hip_bf16.h>

#define QD 256

typedef __attribute__((ext_vector_type(16))) __bf16 v16bf;
typedef __attribute__((ext_vector_type(8)))  float  v8f;
typedef __attribute__((ext_vector_type(2)))  float  v2f;

// ---------------------------------------------------------------- utilities
__global__ __launch_bounds__(256) void zero_f32(float* __restrict__ p, long n) {
  long i = (long)blockIdx.x * blockDim.x + threadIdx.x;
  long stride = (long)gridDim.x * blockDim.x;
  for (; i < n; i += stride) p[i] = 0.0f;
}

__global__ __launch_bounds__(256) void inv_perm_kernel(const int* __restrict__ perm,
                                                       int* __restrict__ inv, int n) {
  int i = blockIdx.x * blockDim.x + threadIdx.x;
  if (i < n) inv[perm[i]] = i;
}

__global__ __launch_bounds__(256) void map_edges_kernel(const int* __restrict__ rows,
                                                        const int* __restrict__ cols,
                                                        const int* __restrict__ inv,
                                                        int* __restrict__ pr,
                                                        int* __restrict__ pc, int e) {
  int i = blockIdx.x * blockDim.x + threadIdx.x;
  if (i < e) { pr[i] = inv[rows[i]]; pc[i] = inv[cols[i]]; }
}

// row-major f32 -> row-major bf16 (layout already [n][k]-contiguous for the GEMM)
__global__ __launch_bounds__(256) void cvt_bf16(const float* __restrict__ src,
                                                __bf16* __restrict__ dst, int n) {
  int i = blockIdx.x * blockDim.x + threadIdx.x;
  if (i < n) dst[i] = (__bf16)src[i];
}

// ---------------------------------------------------------------- SpMM (COO, atomic scatter)
// one wave per edge; lane covers 8 contiguous floats (two float4 loads).
// dense operand [N,256] f32 = 51 MB -> L2-resident on MI455X (192 MB L2),
// so both the gather and the f32 atomic adds stay in L2.
__global__ __launch_bounds__(256) void spmm_coo(const int* __restrict__ r,
                                                const int* __restrict__ c,
                                                const float* __restrict__ v,
                                                const float* __restrict__ X,
                                                float* __restrict__ Y, int E) {
  int wave = threadIdx.x >> 5, lane = threadIdx.x & 31;
  long e = (long)blockIdx.x * 8 + wave;
  if (e >= E) return;
  int rr = r[e], cc = c[e];
  float vv = v[e];
  const float4* src = (const float4*)(X + (long)cc * QD) + lane * 2;
  float4 s0 = src[0], s1 = src[1];
  float* dst = Y + (long)rr * QD + lane * 8;
  atomicAdd(dst + 0, vv * s0.x); atomicAdd(dst + 1, vv * s0.y);
  atomicAdd(dst + 2, vv * s0.z); atomicAdd(dst + 3, vv * s0.w);
  atomicAdd(dst + 4, vv * s1.x); atomicAdd(dst + 5, vv * s1.y);
  atomicAdd(dst + 6, vv * s1.z); atomicAdd(dst + 7, vv * s1.w);
}

// ---------------------------------------------------------------- Gram: G = X^T X via f32 WMMA
// Full f32 precision (keeps G positive-definite for Cholesky) but on the
// matrix pipe: one 16x16 tile of G per block; 8 waves each fold K=4 row
// chunks with V_WMMA_F32_16X16X4_F32, then LDS cross-wave reduction.
// wave id is made scalar via readfirstlane so loop control is SALU-only:
// EXEC stays all-ones through every WMMA (hard ISA requirement).
__global__ __launch_bounds__(256) void gram256_wmma(const float* __restrict__ X,
                                                    float* __restrict__ G, int M) {
  __shared__ float red[8][QD];                 // 8 partial 16x16 tiles
  int wave = __builtin_amdgcn_readfirstlane(threadIdx.x >> 5);  // scalar
  int lane = threadIdx.x & 31;
  int bi = blockIdx.x >> 4, bj = blockIdx.x & 15;
  int h = lane >> 4, ln = lane & 15;
  const float* Ai = X + bi * 16 + ln;          // column (bi*16+ln) of X
  const float* Bj = X + bj * 16 + ln;          // column (bj*16+ln) of X
  v8f acc = {};
  int chunks = M >> 2;                         // M multiple of 4
  for (int t = wave; t < chunks; t += 8) {     // scalar bounds -> s_cbranch loop
    long n0 = (long)t * 4 + 2 * h;
    v2f a, b;
    a[0] = Ai[n0 * QD];
    a[1] = Ai[(n0 + 1) * QD];
    b[0] = Bj[n0 * QD];
    b[1] = Bj[(n0 + 1) * QD];
    acc = __builtin_amdgcn_wmma_f32_16x16x4_f32(false, a, false, b,
                                                (short)0, acc, false, false);
  }
#pragma unroll
  for (int rr = 0; rr < 8; rr++)               // C/D layout: (M=rr+8h, N=ln)
    red[wave][(rr + 8 * h) * 16 + ln] = acc[rr];
  __syncthreads();
  int i = threadIdx.x >> 4, j = threadIdx.x & 15;
  float s = 0.0f;
#pragma unroll
  for (int wv = 0; wv < 8; wv++) s += red[wv][i * 16 + j];
  G[(bi * 16 + i) * QD + (bj * 16 + j)] = s;
}

// ---------------------------------------------------------------- in-place Cholesky (lower), 1 block
__global__ __launch_bounds__(256) void chol256(float* __restrict__ G) {
  int tid = threadIdx.x;
  for (int k = 0; k < QD; k++) {
    if (tid == 0) G[k * QD + k] = sqrtf(fmaxf(G[k * QD + k], 1e-20f));
    __syncthreads();
    float invd = 1.0f / G[k * QD + k];
    if (tid > k) G[tid * QD + k] *= invd;
    __syncthreads();
    if (tid > k) {
      float lik = G[tid * QD + k];
      for (int j = k + 1; j <= tid; j++)
        G[tid * QD + j] = fmaf(-lik, G[j * QD + k], G[tid * QD + j]);
    }
    __syncthreads();
  }
  for (int idx = tid; idx < QD * QD; idx += 256) {
    int i = idx >> 8, j = idx & 255;
    if (j > i) G[idx] = 0.0f;
  }
}

// ---------------------------------------------------------------- Linv (lower tri inverse), 1 block
__global__ __launch_bounds__(256) void triinv256(const float* __restrict__ L,
                                                 float* __restrict__ Li) {
  int j = threadIdx.x;
  for (int i = 0; i < j; i++) Li[i * QD + j] = 0.0f;
  Li[j * QD + j] = 1.0f / L[j * QD + j];
  for (int i = j + 1; i < QD; i++) {
    float s = 0.0f;
    for (int k = j; k < i; k++) s = fmaf(L[i * QD + k], Li[k * QD + j], s);
    Li[i * QD + j] = -s / L[i * QD + i];
  }
}

// ---------------------------------------------------------------- parallel cyclic Jacobi eig, 1 block
__global__ __launch_bounds__(256) void jacobi256(float* __restrict__ M, float* __restrict__ V) {
  int tid = threadIdx.x;
  __shared__ float cs[128], sn[128];
  __shared__ int pp[128], qq[128];
  for (int idx = tid; idx < QD * QD; idx += 256)
    V[idx] = ((idx >> 8) == (idx & 255)) ? 1.0f : 0.0f;
  __syncthreads();
  for (int sweep = 0; sweep < 6; sweep++) {
    for (int r = 0; r < QD - 1; r++) {
      if (tid < 128) {
        int x = (r + tid) % (QD - 1);
        int y = (tid == 0) ? (QD - 1) : (r + QD - 1 - tid) % (QD - 1);
        int p = min(x, y), q = max(x, y);
        pp[tid] = p; qq[tid] = q;
        float app = M[p * QD + p], aqq = M[q * QD + q], apq = M[p * QD + q];
        float c = 1.0f, s = 0.0f;
        if (fabsf(apq) > 1e-12f) {
          float theta = 0.5f * (aqq - app) / apq;
          float t = copysignf(1.0f, theta) / (fabsf(theta) + sqrtf(fmaf(theta, theta, 1.0f)));
          c = 1.0f / sqrtf(fmaf(t, t, 1.0f));
          s = t * c;
        }
        cs[tid] = c; sn[tid] = s;
      }
      __syncthreads();
      for (int w = tid; w < 128 * QD; w += 256) {     // M*J and V*J (columns)
        int pi = w >> 8, row = w & 255;
        int p = pp[pi], q = qq[pi];
        float c = cs[pi], s = sn[pi];
        float mp = M[row * QD + p], mq = M[row * QD + q];
        M[row * QD + p] = fmaf(c, mp, -s * mq);
        M[row * QD + q] = fmaf(s, mp,  c * mq);
        float vp = V[row * QD + p], vq = V[row * QD + q];
        V[row * QD + p] = fmaf(c, vp, -s * vq);
        V[row * QD + q] = fmaf(s, vp,  c * vq);
      }
      __syncthreads();
      for (int w = tid; w < 128 * QD; w += 256) {     // J^T * M (rows)
        int pi = w >> 8, col = w & 255;
        int p = pp[pi], q = qq[pi];
        float c = cs[pi], s = sn[pi];
        float mp = M[p * QD + col], mq = M[q * QD + col];
        M[p * QD + col] = fmaf(c, mp, -s * mq);
        M[q * QD + col] = fmaf(s, mp,  c * mq);
      }
      __syncthreads();
    }
  }
}

// sort eigenpairs descending; emit Ub pre-transposed and pre-converted:
// Bbf[n*QD + k] = Ub[k][n] = V[k][ord[n]]  (exactly the GEMM's B layout)
__global__ __launch_bounds__(256) void sort_ub_bf16(const float* __restrict__ M,
                                                    const float* __restrict__ V,
                                                    __bf16* __restrict__ Bbf) {
  __shared__ float ev[QD];
  __shared__ int ord[QD];
  int tid = threadIdx.x;
  ev[tid] = M[tid * QD + tid];
  ord[tid] = tid;
  __syncthreads();
  if (tid == 0) {
    for (int i = 0; i < QD - 1; i++) {
      int mx = i;
      for (int j = i + 1; j < QD; j++)
        if (ev[ord[j]] > ev[ord[mx]]) mx = j;
      int t = ord[i]; ord[i] = ord[mx]; ord[mx] = t;
    }
  }
  __syncthreads();
  for (int idx = tid; idx < QD * QD; idx += 256) {
    int n = idx >> 8, k = idx & 255;
    Bbf[idx] = (__bf16)V[k * QD + ord[n]];
  }
}

// ---------------------------------------------------------------- WMMA bf16 GEMM, K = N = 256
// C[M,256] = op(A[M,256]) @ B (+bias+relu). B is pre-converted bf16 in
// [n][k]-contiguous layout -> per-lane 16-element runs are one aligned
// 32-byte load. Wave computes a 16x64 strip (4 accumulators, A reused 4x);
// block = 128 threads = 4 waves covering N=256; grid = M/16 exactly ->
// every wave fully active (EXEC all-ones for WMMA).
template <bool GATHER, bool EPI>
__global__ __launch_bounds__(128) void gemm_wmma(const float* __restrict__ A,
                                                 const int* __restrict__ rowidx,
                                                 const __bf16* __restrict__ Bbf,
                                                 const float* __restrict__ bias,
                                                 float* __restrict__ C) {
  int wave = __builtin_amdgcn_readfirstlane(threadIdx.x >> 5);  // scalar n-quad
  int lane = threadIdx.x & 31;
  int mt = blockIdx.x;
  int h = lane >> 4, ln = lane & 15;

  int mrow = mt * 16 + ln;
  int arow = GATHER ? rowidx[mrow] : mrow;
  const float* Abase = A + (long)arow * QD;
  __builtin_prefetch(Abase, 0, 3);      // global_prefetch_b8: pull A row toward L0/L2

  int nbase = wave * 64;
  v8f acc[4];
#pragma unroll
  for (int j = 0; j < 4; j++) {
    float binit = EPI ? bias[nbase + j * 16 + ln] : 0.0f;
#pragma unroll
    for (int rr = 0; rr < 8; rr++) acc[j][rr] = binit;
  }

#pragma unroll
  for (int kk = 0; kk < 8; kk++) {
    int k0 = kk * 32;
    // A fragment: lane half h holds K = k0+16h+e (e<16)
    float4 a0 = *(const float4*)(Abase + k0 + 16 * h);
    float4 a1 = *(const float4*)(Abase + k0 + 16 * h + 4);
    float4 a2 = *(const float4*)(Abase + k0 + 16 * h + 8);
    float4 a3 = *(const float4*)(Abase + k0 + 16 * h + 12);
    v16bf a;
    a[0]  = (__bf16)a0.x; a[1]  = (__bf16)a0.y; a[2]  = (__bf16)a0.z; a[3]  = (__bf16)a0.w;
    a[4]  = (__bf16)a1.x; a[5]  = (__bf16)a1.y; a[6]  = (__bf16)a1.z; a[7]  = (__bf16)a1.w;
    a[8]  = (__bf16)a2.x; a[9]  = (__bf16)a2.y; a[10] = (__bf16)a2.z; a[11] = (__bf16)a2.w;
    a[12] = (__bf16)a3.x; a[13] = (__bf16)a3.y; a[14] = (__bf16)a3.z; a[15] = (__bf16)a3.w;
#pragma unroll
    for (int j = 0; j < 4; j++) {
      // B fragment: lanes 0-15 hold K = k0..k0+15, lanes 16-31 hold K = k0+16..k0+31
      const __bf16* bp = Bbf + (long)(nbase + j * 16 + ln) * QD + k0 + 16 * h;
      v16bf b = *(const v16bf*)bp;     // one aligned 32B load (2x b128)
      acc[j] = __builtin_amdgcn_wmma_f32_16x16x32_bf16(false, a, false, b,
                                                       (short)0, acc[j], false, false);
    }
  }

#pragma unroll
  for (int j = 0; j < 4; j++) {
#pragma unroll
    for (int rr = 0; rr < 8; rr++) {
      float vv = acc[j][rr];
      if (EPI) vv = fmaxf(vv, 0.0f);
      C[(long)(mt * 16 + rr + 8 * h) * QD + nbase + j * 16 + ln] = vv;
    }
  }
}

// ---------------------------------------------------------------- host orchestration
static inline void qr_step(float* X, float* Gm, float* Li, __bf16* Bbf,
                           float* Qout, int N, hipStream_t s) {
  gram256_wmma<<<256, 256, 0, s>>>(X, Gm, N);
  chol256<<<1, 256, 0, s>>>(Gm);
  triinv256<<<1, 256, 0, s>>>(Gm, Li);
  cvt_bf16<<<QD * QD / 256, 256, 0, s>>>(Li, Bbf, QD * QD);   // B[n][k] = Linv[n][k] = (L^-T)[k][n]
  gemm_wmma<false, false><<<N / 16, 128, 0, s>>>(X, nullptr, Bbf, nullptr, Qout);
}

extern "C" void kernel_launch(void* const* d_in, const int* in_sizes, int n_in,
                              void* d_out, int out_size, void* d_ws, size_t ws_size,
                              hipStream_t stream) {
  const int*   rows  = (const int*)  d_in[1];
  const int*   cols  = (const int*)  d_in[2];
  const float* vals  = (const float*)d_in[3];
  const int*   perm  = (const int*)  d_in[4];
  const float* omega = (const float*)d_in[5];
  const float* W1    = (const float*)d_in[6];
  const float* b1    = (const float*)d_in[7];
  const float* W2    = (const float*)d_in[8];
  const float* b2    = (const float*)d_in[9];
  float* out = (float*)d_out;

  const int N = in_sizes[4];   // perm length = num nodes
  const int E = in_sizes[1];   // rows length = num edges
  const long NQ = (long)N * QD;

  // workspace carve-out (keep 32B alignment for the bf16 vector loads)
  char* w = (char*)d_ws;
  float*  buf0 = (float*)w;  w += NQ * sizeof(float);
  float*  buf1 = (float*)w;  w += NQ * sizeof(float);
  float*  Gm   = (float*)w;  w += QD * QD * sizeof(float);
  float*  Li   = (float*)w;  w += QD * QD * sizeof(float);
  float*  Vm   = (float*)w;  w += QD * QD * sizeof(float);
  __bf16* Bbf  = (__bf16*)w; w += QD * QD * sizeof(__bf16);
  int*    invp = (int*)w;    w += (long)N * sizeof(int);
  int*    pr   = (int*)w;    w += (long)E * sizeof(int);
  int*    pc   = (int*)w;    w += (long)E * sizeof(int);

  int nb256 = (N + 255) / 256;
  int eb256 = (E + 255) / 256;
  int spmmB = (E + 7) / 8;      // one wave per edge, 8 waves/block
  int zeroB = 2048;

  // 1. inverse permutation + permuted COO indices
  inv_perm_kernel<<<nb256, 256, 0, stream>>>(perm, invp, N);
  map_edges_kernel<<<eb256, 256, 0, stream>>>(rows, cols, invp, pr, pc, E);

  // 2. X1 = A' @ Omega ; Q1 = cholQR(X1)
  zero_f32<<<zeroB, 256, 0, stream>>>(buf0, NQ);
  spmm_coo<<<spmmB, 256, 0, stream>>>(pr, pc, vals, omega, buf0, E);
  qr_step(buf0, Gm, Li, Bbf, buf1, N, stream);                 // Q1 -> buf1

  // 3. X2 = A'^T @ Q1 ; Q2 = cholQR(X2)
  zero_f32<<<zeroB, 256, 0, stream>>>(buf0, NQ);
  spmm_coo<<<spmmB, 256, 0, stream>>>(pc, pr, vals, buf1, buf0, E);
  qr_step(buf0, Gm, Li, Bbf, buf1, N, stream);                 // Q2 -> buf1

  // 4. X3 = A' @ Q2 ; Q3 = cholQR(X3)  (this is Qb)
  zero_f32<<<zeroB, 256, 0, stream>>>(buf0, NQ);
  spmm_coo<<<spmmB, 256, 0, stream>>>(pr, pc, vals, buf1, buf0, E);
  qr_step(buf0, Gm, Li, Bbf, buf1, N, stream);                 // Q3 = Qb -> buf1

  // 5. Y = A'^T @ Qb (= B^T) ; M = Y^T Y = B B^T ; eig(M) -> Ub (pre-transposed bf16)
  zero_f32<<<zeroB, 256, 0, stream>>>(buf0, NQ);
  spmm_coo<<<spmmB, 256, 0, stream>>>(pc, pr, vals, buf1, buf0, E);
  gram256_wmma<<<256, 256, 0, stream>>>(buf0, Gm, N);
  jacobi256<<<1, 256, 0, stream>>>(Gm, Vm);
  sort_ub_bf16<<<1, 256, 0, stream>>>(Gm, Vm, Bbf);

  // 6. U = Qb @ Ub
  gemm_wmma<false, false><<<N / 16, 128, 0, stream>>>(buf1, nullptr, Bbf, nullptr, buf0);

  // 7. emb = relu(U[inv_perm] @ W1^T + b1)   (gather fused into A-load; W1 row-major is already [n][k])
  cvt_bf16<<<QD * QD / 256, 256, 0, stream>>>(W1, Bbf, QD * QD);
  gemm_wmma<true, true><<<N / 16, 128, 0, stream>>>(buf0, invp, Bbf, b1, buf1);

  // 8. out = relu(emb @ W2^T + b2)
  cvt_bf16<<<QD * QD / 256, 256, 0, stream>>>(W2, Bbf, QD * QD);
  gemm_wmma<false, true><<<N / 16, 128, 0, stream>>>(buf1, nullptr, Bbf, b2, out);

  (void)n_in; (void)out_size; (void)ws_size;
}
